// CrossMamba_27668179320991
// MI455X (gfx1250) — compile-verified, matrix operand order checked by reading, not run on hardware
//
#include <hip/hip_runtime.h>
#include <hip/hip_bf16.h>
#include <math.h>
#include <stdint.h>

// ---------------- problem constants (from reference) ----------------
#define BB     4
#define FEATD  64
#define CCH    128
#define LLEN   (FEATD*FEATD)      // 4096
#define DIN    256                // 2*C
#define DSTATE 16
#define DTR    8
#define DCONV  4
#define ROWS   (BB*LLEN)          // 16384
#define NXZ    (2*DIN)            // 512
#define NDBL   (DTR + 2*DSTATE)   // 40
#define C2     (2*CCH)            // 256
#define TCH    64                 // scan chunk length
#define NCHUNK (LLEN/TCH)         // 64

typedef __attribute__((ext_vector_type(16))) _Float16 v16h;
typedef __attribute__((ext_vector_type(8)))  float    v8f;

#if defined(__HIP_DEVICE_COMPILE__) && __has_builtin(__builtin_amdgcn_tensor_load_to_lds)
#define HAVE_TDM 1
typedef __attribute__((ext_vector_type(4))) unsigned int u32x4;
typedef __attribute__((ext_vector_type(8))) int          i32x8;
typedef __attribute__((ext_vector_type(4))) int          i32x4;
#endif

// ---------------- WMMA A-tile load (wave32, 16x16x32 f16) from global ----------------
// lanes 0-15: row = lane, K = {0..7, 16..23}; lanes 16-31: row = lane-16, K = {8..15, 24..31}
__device__ __forceinline__ v16h load_tile_16x32(const _Float16* __restrict__ A, int lda, int lane) {
  int row = lane & 15;
  int kb  = (lane >> 4) << 3;            // 0 or 8
  const _Float16* p = A + (size_t)row * lda + kb;
  v16h a;
#pragma unroll
  for (int i = 0; i < 8; ++i) a[i] = p[i];
#pragma unroll
  for (int i = 0; i < 8; ++i) a[8 + i] = p[16 + i];
  return a;
}

// ---------------- kernel 1: res = rgb+residual, LN -> u_h ; LN(depth) -> flipped e_h ----------------
__global__ void k_add_ln(const float* __restrict__ rgb, const float* __restrict__ resid,
                         const float* __restrict__ depth,
                         const float* __restrict__ sn1w, const float* __restrict__ sn1b,
                         const float* __restrict__ sn2w, const float* __restrict__ sn2b,
                         float* __restrict__ res, _Float16* __restrict__ u_h,
                         _Float16* __restrict__ e_h) {
  int r = blockIdx.x;                 // row in [0, ROWS)
  int c = threadIdx.x;                // channel in [0,128)
  int b = r / LLEN, l = r % LLEN;
  __shared__ float s1[CCH];

  float v = rgb[(size_t)r*CCH + c] + resid[(size_t)r*CCH + c];
  res[(size_t)r*CCH + c] = v;

  s1[c] = v; __syncthreads();
  for (int off = 64; off > 0; off >>= 1) { if (c < off) s1[c] += s1[c + off]; __syncthreads(); }
  float mu = s1[0] * (1.0f / CCH); __syncthreads();
  float d0 = v - mu;
  s1[c] = d0 * d0; __syncthreads();
  for (int off = 64; off > 0; off >>= 1) { if (c < off) s1[c] += s1[c + off]; __syncthreads(); }
  float var = s1[0] * (1.0f / CCH); __syncthreads();
  u_h[(size_t)r*CCH + c] = (_Float16)(d0 * rsqrtf(var + 1e-5f) * sn1w[c] + sn1b[c]);

  float dv = depth[(size_t)r*CCH + c];
  s1[c] = dv; __syncthreads();
  for (int off = 64; off > 0; off >>= 1) { if (c < off) s1[c] += s1[c + off]; __syncthreads(); }
  float mu2 = s1[0] * (1.0f / CCH); __syncthreads();
  float d2 = dv - mu2;
  s1[c] = d2 * d2; __syncthreads();
  for (int off = 64; off > 0; off >>= 1) { if (c < off) s1[c] += s1[c + off]; __syncthreads(); }
  float var2 = s1[0] * (1.0f / CCH);
  int rf = b*LLEN + (LLEN - 1 - l);
  e_h[(size_t)rf*CCH + c] = (_Float16)(d2 * rsqrtf(var2 + 1e-5f) * sn2w[c] + sn2b[c]);
}

// ---------------- f32 -> f16 converter (weights) ----------------
__global__ void k_tohalf(const float* __restrict__ s, _Float16* __restrict__ d, int n) {
  int i = blockIdx.x * blockDim.x + threadIdx.x;
  if (i < n) d[i] = (_Float16)s[i];
}

// =====================================================================
// LDS-staged WMMA GEMMs: one block = 8 waves sharing one 16-wide N tile.
// Block stages W tile (16 x K f16) in LDS once; waves cover 8 M-tiles.
// =====================================================================

// ---------------- GEMM: xz = A(16384x128) x W^T(128x512), out f16 ----------------
__global__ void k_gemm_xz(const _Float16* __restrict__ A, const _Float16* __restrict__ W,
                          _Float16* __restrict__ out) {
  const int ntn = NXZ / 16;                 // 32
  __shared__ _Float16 wt[16 * CCH];         // 4 KB
  int tn   = blockIdx.x % ntn;
  int mblk = blockIdx.x / ntn;              // 0..ROWS/128-1
  const _Float16* Wtile = W + (size_t)tn * 16 * CCH;

#if defined(HAVE_TDM)
  // Tensor Data Mover: async DMA of the 16xK f16 weight tile into LDS.
  // D# built per CDNA5 ISA (group0: count/lds/global/type, group1: dims/tile).
  {
    unsigned long long ga = (unsigned long long)(uintptr_t)Wtile;     // byte addr
    unsigned int lds = (unsigned int)(uintptr_t)(void*)wt;            // LDS byte addr
    u32x4 g0 = { 1u,                                                   // count=1 (valid)
                 lds,
                 (unsigned int)ga,
                 (unsigned int)((ga >> 32) & 0x01FFFFFFu) | (2u << 30) }; // addr[56:32] | type=2
    i32x8 g1 = { (int)0x00010000,          // data_size=1 (2 bytes/elem)
                 (int)((CCH & 0xFFFF) << 16),          // tensor_dim0 lo16
                 (int)(((unsigned)CCH >> 16) | ((NXZ & 0xFFFF) << 16)), // dim0 hi | dim1 lo
                 (int)(((unsigned)NXZ >> 16) | ((CCH & 0xFFFF) << 16)), // dim1 hi | tile_dim0
                 16,                        // tile_dim1 = 16 rows (tile_dim2 = 0)
                 CCH,                       // tensor_dim0_stride lo32 (elements)
                 0, 0 };
    i32x4 gz = { 0, 0, 0, 0 };
    if (threadIdx.x < 32) {                 // one wave issues the DMA
#if defined(__clang_major__) && (__clang_major__ >= 23)
      i32x8 gpad = { 0, 0, 0, 0, 0, 0, 0, 0 };
      __builtin_amdgcn_tensor_load_to_lds(g0, g1, gz, gz, gpad, 0);
#else
      __builtin_amdgcn_tensor_load_to_lds(g0, g1, gz, gz, 0);
#endif
      __builtin_amdgcn_s_wait_tensorcnt(0);
    }
  }
  __syncthreads();
#endif
  // cooperative stage of W tile (authoritative contents): 16*128 halves = 256 x int4
  {
    const int4* src = (const int4*)Wtile;
    int4* dst = (int4*)wt;
    dst[threadIdx.x] = src[threadIdx.x];
  }
  __syncthreads();

  int wave = threadIdx.x >> 5;
  int lane = threadIdx.x & 31;
  int tm = mblk * 8 + wave;
  const _Float16* Ap = A + (size_t)tm * 16 * CCH;
  __builtin_prefetch(Ap, 0, 3);             // global_prefetch_b8 on the A strip
  int wrow = lane & 15, kb = (lane >> 4) << 3;
  v8f acc = {};
#pragma unroll
  for (int k0 = 0; k0 < CCH; k0 += 32) {
    v16h a = load_tile_16x32(Ap + k0, CCH, lane);
    v16h bm;
#pragma unroll
    for (int i = 0; i < 8; ++i) bm[i] = wt[wrow * CCH + k0 + kb + i];
#pragma unroll
    for (int i = 0; i < 8; ++i) bm[8 + i] = wt[wrow * CCH + k0 + 16 + kb + i];
    acc = __builtin_amdgcn_wmma_f32_16x16x32_f16(false, a, false, bm, (short)0, acc, false, false);
  }
  int rb = tm*16 + ((lane >> 4) << 3);
  int cb = tn*16 + (lane & 15);
#pragma unroll
  for (int v = 0; v < 8; ++v)
    out[(size_t)(rb + v) * NXZ + cb] = (_Float16)acc[v];
}

// ---------------- causal depthwise conv (k=4) + SiLU ----------------
__global__ void k_dwconv(const _Float16* __restrict__ xz, const float* __restrict__ cw,
                         const float* __restrict__ cb, _Float16* __restrict__ xc) {
  int idx = blockIdx.x * blockDim.x + threadIdx.x;
  if (idx >= ROWS * DIN) return;
  int d = idx % DIN, r = idx / DIN;
  int b = r / LLEN, l = r % LLEN;
  float acc = cb[d];
#pragma unroll
  for (int i = 0; i < DCONV; ++i) {
    int li = l + i - (DCONV - 1);
    if (li >= 0) acc += (float)xz[((size_t)b*LLEN + li) * NXZ + d] * cw[d*DCONV + i];
  }
  float s = acc / (1.f + expf(-acc));             // silu
  xc[(size_t)r * DIN + d] = (_Float16)s;
}

// ---------------- dbl = xc @ xproj.T  (N=40, scalar) ----------------
__global__ void k_dbl(const _Float16* __restrict__ xc, const float* __restrict__ xproj,
                      float* __restrict__ dbl) {
  int idx = blockIdx.x * blockDim.x + threadIdx.x;
  if (idx >= ROWS * NDBL) return;
  int j = idx % NDBL, r = idx / NDBL;
  const _Float16* xp = xc + (size_t)r * DIN;
  const float* wp = xproj + (size_t)j * DIN;
  float acc = 0.f;
#pragma unroll 8
  for (int k = 0; k < DIN; ++k) acc += (float)xp[k] * wp[k];
  dbl[(size_t)r * NDBL + j] = acc;
}

// ---------------- delta = softplus(dt @ dt_w.T + dt_b) ----------------
__global__ void k_delta(const float* __restrict__ dbl, const float* __restrict__ dt_w,
                        const float* __restrict__ dt_b, float* __restrict__ delta) {
  int idx = blockIdx.x * blockDim.x + threadIdx.x;
  if (idx >= ROWS * DIN) return;
  int d = idx % DIN, r = idx / DIN;
  float acc = dt_b[d];
  const float* dt = dbl + (size_t)r * NDBL;
#pragma unroll
  for (int k = 0; k < DTR; ++k) acc += dt_w[d*DTR + k] * dt[k];
  delta[idx] = (acc > 0.f) ? (acc + log1pf(expf(-acc))) : log1pf(expf(acc));
}

// =====================================================================
// Chunk-parallel selective scan: h_t = a_t h_{t-1} + u_t.
// Pass1: per-chunk (P = prod a, S = local scan).  Pass2: 64-step serial
// carry over chunks.  Pass3: replay with h-start.  Depth 4096 -> ~64.
// =====================================================================

__global__ void k_scan_p1(const float* __restrict__ dbl, const float* __restrict__ delta,
                          const _Float16* __restrict__ xc, const float* __restrict__ A_log,
                          float* __restrict__ Pc, float* __restrict__ Sc) {
  int b = blockIdx.x;
  int c = blockIdx.y;
  int d = threadIdx.x;
  float Aa[DSTATE];
#pragma unroll
  for (int s = 0; s < DSTATE; ++s) Aa[s] = -expf(A_log[d*DSTATE + s]);
  float P[DSTATE], S[DSTATE];
#pragma unroll
  for (int s = 0; s < DSTATE; ++s) { P[s] = 1.f; S[s] = 0.f; }
  __shared__ float sB[DSTATE];
  for (int tt = 0; tt < TCH; ++tt) {
    size_t row = (size_t)b * LLEN + (size_t)c * TCH + tt;
    if (d < DSTATE) sB[d] = dbl[row * NDBL + DTR + d];
    __syncthreads();
    float dl = delta[row * DIN + d];
    float dx = dl * (float)xc[row * DIN + d];
#pragma unroll
    for (int s = 0; s < DSTATE; ++s) {
      float a = expf(dl * Aa[s]);
      P[s] *= a;
      S[s] = a * S[s] + dx * sB[s];
    }
    __syncthreads();
  }
  size_t base = (((size_t)c * BB + b) * DIN + d) * DSTATE;
#pragma unroll
  for (int s = 0; s < DSTATE; ++s) { Pc[base + s] = P[s]; Sc[base + s] = S[s]; }
}

__global__ void k_scan_p2(const float* __restrict__ Pc, float* __restrict__ Sc) {
  int idx = blockIdx.x * blockDim.x + threadIdx.x;      // (b*DIN + d)*DSTATE + s
  if (idx >= BB * DIN * DSTATE) return;
  const size_t stride = (size_t)BB * DIN * DSTATE;
  float h = 0.f;
  for (int c = 0; c < NCHUNK; ++c) {
    size_t i = (size_t)c * stride + idx;
    float P = Pc[i], S = Sc[i];
    Sc[i] = h;                 // h at chunk entry
    h = P * h + S;
  }
}

__global__ void k_scan_p3(const float* __restrict__ dbl, const float* __restrict__ delta,
                          const _Float16* __restrict__ xc, const _Float16* __restrict__ xz,
                          const float* __restrict__ A_log, const float* __restrict__ Dp,
                          const float* __restrict__ Hs, float* __restrict__ yg) {
  int b = blockIdx.x;
  int c = blockIdx.y;
  int d = threadIdx.x;
  float Aa[DSTATE];
#pragma unroll
  for (int s = 0; s < DSTATE; ++s) Aa[s] = -expf(A_log[d*DSTATE + s]);
  float Dd = Dp[d];
  float h[DSTATE];
  size_t base = (((size_t)c * BB + b) * DIN + d) * DSTATE;
#pragma unroll
  for (int s = 0; s < DSTATE; ++s) h[s] = Hs[base + s];
  __shared__ float sBC[2 * DSTATE];
  for (int tt = 0; tt < TCH; ++tt) {
    size_t row = (size_t)b * LLEN + (size_t)c * TCH + tt;
    if (d < 2*DSTATE) sBC[d] = dbl[row * NDBL + DTR + d];
    __syncthreads();
    float dl = delta[row * DIN + d];
    float xt = (float)xc[row * DIN + d];
    float dx = dl * xt;
    float y = 0.f;
#pragma unroll
    for (int s = 0; s < DSTATE; ++s) {
      h[s] = expf(dl * Aa[s]) * h[s] + dx * sBC[s];
      y += h[s] * sBC[DSTATE + s];
    }
    y += xt * Dd;
    float z = (float)xz[row * NXZ + DIN + d];
    yg[row * DIN + d] = y * (z / (1.f + expf(-z)));
    __syncthreads();
  }
}

// ---------------- ysum = y_main + flip(y_extra), f16 ----------------
__global__ void k_ysum(const float* __restrict__ ym, const float* __restrict__ ye,
                       _Float16* __restrict__ ys) {
  int idx = blockIdx.x * blockDim.x + threadIdx.x;
  if (idx >= ROWS * DIN) return;
  int d = idx % DIN, r = idx / DIN;
  int b = r / LLEN, l = r % LLEN;
  size_t rf = (size_t)b * LLEN + (LLEN - 1 - l);
  ys[(size_t)r * DIN + d] = (_Float16)(ym[(size_t)r * DIN + d] + ye[rf * DIN + d]);
}

// ---------------- GEMM: gf = ysum(16384x256) x out_w^T(256x128), +res, scatter NCHW ----------------
__global__ void k_gemm_gf(const _Float16* __restrict__ A, const _Float16* __restrict__ W,
                          const float* __restrict__ res, float* __restrict__ spatial) {
  const int ntn = CCH / 16;                 // 8
  __shared__ _Float16 wt[16 * DIN];         // 8 KB
  int tn   = blockIdx.x % ntn;
  int mblk = blockIdx.x / ntn;
  {
    const int4* src = (const int4*)(W + (size_t)tn * 16 * DIN);
    int4* dst = (int4*)wt;
    dst[threadIdx.x]       = src[threadIdx.x];
    dst[threadIdx.x + 256] = src[threadIdx.x + 256];
  }
  __syncthreads();

  int wave = threadIdx.x >> 5;
  int lane = threadIdx.x & 31;
  int tm = mblk * 8 + wave;
  const _Float16* Ap = A + (size_t)tm * 16 * DIN;
  __builtin_prefetch(Ap, 0, 3);
  int wrow = lane & 15, kb = (lane >> 4) << 3;
  v8f acc = {};
#pragma unroll
  for (int k0 = 0; k0 < DIN; k0 += 32) {
    v16h a = load_tile_16x32(Ap + k0, DIN, lane);
    v16h bm;
#pragma unroll
    for (int i = 0; i < 8; ++i) bm[i] = wt[wrow * DIN + k0 + kb + i];
#pragma unroll
    for (int i = 0; i < 8; ++i) bm[8 + i] = wt[wrow * DIN + k0 + 16 + kb + i];
    acc = __builtin_amdgcn_wmma_f32_16x16x32_f16(false, a, false, bm, (short)0, acc, false, false);
  }
  int rb = tm*16 + ((lane >> 4) << 3);
  int cc = tn*16 + (lane & 15);
#pragma unroll
  for (int v = 0; v < 8; ++v) {
    int r = rb + v;
    int b = r / LLEN, p = r % LLEN;
    float val = acc[v] + res[(size_t)r * CCH + cc];
    spatial[((size_t)b * CCH + cc) * LLEN + p] = val;    // NCHW
  }
}

// ---------------- LayerNorm2d over channels per pixel -> f16 NHWC ----------------
__global__ void k_ln2d(const float* __restrict__ sp, const float* __restrict__ cw,
                       const float* __restrict__ cb, _Float16* __restrict__ lnh) {
  int pix = blockIdx.x;                // b*LLEN + p
  int c = threadIdx.x;                 // 0..127
  int b = pix / LLEN, p = pix % LLEN;
  __shared__ float s[CCH];
  float v = sp[((size_t)b * CCH + c) * LLEN + p];
  s[c] = v; __syncthreads();
  for (int off = 64; off > 0; off >>= 1) { if (c < off) s[c] += s[c + off]; __syncthreads(); }
  float mu = s[0] * (1.0f / CCH); __syncthreads();
  float d = v - mu;
  s[c] = d * d; __syncthreads();
  for (int off = 64; off > 0; off >>= 1) { if (c < off) s[c] += s[c + off]; __syncthreads(); }
  float var = s[0] * (1.0f / CCH);
  lnh[(size_t)pix * CCH + c] = (_Float16)(cw[c] * (d * rsqrtf(var + 1e-6f)) + cb[c]);
}

// ---------------- GEMM f1: (16384x128) x f1_w^T(128x256) + bias -> f32 NHWC ----------------
__global__ void k_gemm_f1(const _Float16* __restrict__ A, const _Float16* __restrict__ W,
                          const float* __restrict__ bias, float* __restrict__ out) {
  const int ntn = C2 / 16;                  // 16
  __shared__ _Float16 wt[16 * CCH];         // 4 KB
  int tn   = blockIdx.x % ntn;
  int mblk = blockIdx.x / ntn;
  {
    const int4* src = (const int4*)(W + (size_t)tn * 16 * CCH);
    int4* dst = (int4*)wt;
    dst[threadIdx.x] = src[threadIdx.x];
  }
  __syncthreads();

  int wave = threadIdx.x >> 5;
  int lane = threadIdx.x & 31;
  int tm = mblk * 8 + wave;
  const _Float16* Ap = A + (size_t)tm * 16 * CCH;
  __builtin_prefetch(Ap, 0, 3);
  int wrow = lane & 15, kb = (lane >> 4) << 3;
  v8f acc = {};
#pragma unroll
  for (int k0 = 0; k0 < CCH; k0 += 32) {
    v16h a = load_tile_16x32(Ap + k0, CCH, lane);
    v16h bm;
#pragma unroll
    for (int i = 0; i < 8; ++i) bm[i] = wt[wrow * CCH + k0 + kb + i];
#pragma unroll
    for (int i = 0; i < 8; ++i) bm[8 + i] = wt[wrow * CCH + k0 + 16 + kb + i];
    acc = __builtin_amdgcn_wmma_f32_16x16x32_f16(false, a, false, bm, (short)0, acc, false, false);
  }
  int rb = tm*16 + ((lane >> 4) << 3);
  int c  = tn*16 + (lane & 15);
  float bv = bias[c];
#pragma unroll
  for (int v = 0; v < 8; ++v)
    out[(size_t)(rb + v) * C2 + c] = acc[v] + bv;
}

// ---------------- grouped 3x3 conv (groups=128, 2-in/2-out per group), pad=1 ----------------
__global__ void k_gconv(const float* __restrict__ h1, const float* __restrict__ w2,
                        const float* __restrict__ b2, _Float16* __restrict__ h2) {
  int idx = blockIdx.x * blockDim.x + threadIdx.x;
  if (idx >= ROWS * C2) return;
  int oc = idx % C2, pix = idx / C2;
  int b = pix / LLEN, p = pix % LLEN;
  int py = p >> 6, px = p & 63;
  int g = oc >> 1;
  float acc = b2[oc];
#pragma unroll
  for (int ic2 = 0; ic2 < 2; ++ic2) {
    int ic = 2*g + ic2;
    const float* wk = w2 + ((size_t)oc * 2 + ic2) * 9;
#pragma unroll
    for (int ky = 0; ky < 3; ++ky) {
      int yy = py + ky - 1; if (yy < 0 || yy >= FEATD) continue;
#pragma unroll
      for (int kx = 0; kx < 3; ++kx) {
        int xx = px + kx - 1; if (xx < 0 || xx >= FEATD) continue;
        acc += h1[(((size_t)b * LLEN) + yy*FEATD + xx) * C2 + ic] * wk[ky*3 + kx];
      }
    }
  }
  h2[(size_t)pix * C2 + oc] = (_Float16)acc;
}

// ---------------- GEMM f3: (16384x256) x f3_w^T(256x128) + bias -> scatter NCHW into d_out ----------------
__global__ void k_gemm_f3(const _Float16* __restrict__ A, const _Float16* __restrict__ W,
                          const float* __restrict__ bias, float* __restrict__ out) {
  const int ntn = CCH / 16;                 // 8
  __shared__ _Float16 wt[16 * C2];          // 8 KB
  int tn   = blockIdx.x % ntn;
  int mblk = blockIdx.x / ntn;
  {
    const int4* src = (const int4*)(W + (size_t)tn * 16 * C2);
    int4* dst = (int4*)wt;
    dst[threadIdx.x]       = src[threadIdx.x];
    dst[threadIdx.x + 256] = src[threadIdx.x + 256];
  }
  __syncthreads();

  int wave = threadIdx.x >> 5;
  int lane = threadIdx.x & 31;
  int tm = mblk * 8 + wave;
  const _Float16* Ap = A + (size_t)tm * 16 * C2;
  __builtin_prefetch(Ap, 0, 3);
  int wrow = lane & 15, kb = (lane >> 4) << 3;
  v8f acc = {};
#pragma unroll
  for (int k0 = 0; k0 < C2; k0 += 32) {
    v16h a = load_tile_16x32(Ap + k0, C2, lane);
    v16h bm;
#pragma unroll
    for (int i = 0; i < 8; ++i) bm[i] = wt[wrow * C2 + k0 + kb + i];
#pragma unroll
    for (int i = 0; i < 8; ++i) bm[8 + i] = wt[wrow * C2 + k0 + 16 + kb + i];
    acc = __builtin_amdgcn_wmma_f32_16x16x32_f16(false, a, false, bm, (short)0, acc, false, false);
  }
  int rb = tm*16 + ((lane >> 4) << 3);
  int c  = tn*16 + (lane & 15);
  float bv = bias[c];
#pragma unroll
  for (int v = 0; v < 8; ++v) {
    int r = rb + v;
    int b = r / LLEN, p = r % LLEN;
    out[((size_t)b * CCH + c) * LLEN + p] = acc[v] + bv;
  }
}

// ---------------- copy res into second output slot ----------------
__global__ void k_copy(const float* __restrict__ src, float* __restrict__ dst, int n) {
  int i = blockIdx.x * blockDim.x + threadIdx.x;
  if (i < n) dst[i] = src[i];
}

// ====================================================================
extern "C" void kernel_launch(void* const* d_in, const int* in_sizes, int n_in,
                              void* d_out, int out_size, void* d_ws, size_t ws_size,
                              hipStream_t stream) {
  (void)in_sizes; (void)n_in; (void)out_size; (void)ws_size;

  const float* rgb      = (const float*)d_in[0];
  const float* residual = (const float*)d_in[1];
  const float* depth    = (const float*)d_in[2];
  // d_in[3] = feat_dim (constant 64, compiled in)
  const float* sn1w = (const float*)d_in[4];
  const float* sn1b = (const float*)d_in[5];
  const float* sn2w = (const float*)d_in[6];
  const float* sn2b = (const float*)d_in[7];
  const float* cnw  = (const float*)d_in[8];
  const float* cnb  = (const float*)d_in[9];
  const float* out_w = (const float*)d_in[10];
  const float* f1w = (const float*)d_in[11];
  const float* f1b = (const float*)d_in[12];
  const float* f2w = (const float*)d_in[13];
  const float* f2b = (const float*)d_in[14];
  const float* f3w = (const float*)d_in[15];
  const float* f3b = (const float*)d_in[16];

  // ---- workspace carve-out ----
  char* ws = (char*)d_ws;
  size_t off = 0;
  auto carve = [&](size_t bytes) -> void* {
    void* p = ws + off;
    off += (bytes + 255) & ~(size_t)255;
    return p;
  };
  float*    res    = (float*)   carve((size_t)ROWS * CCH * 4);
  _Float16* u_h    = (_Float16*)carve((size_t)ROWS * CCH * 2);
  _Float16* e_h    = (_Float16*)carve((size_t)ROWS * CCH * 2);
  _Float16* inw_h0 = (_Float16*)carve((size_t)NXZ * CCH * 2);
  _Float16* inw_h1 = (_Float16*)carve((size_t)NXZ * CCH * 2);
  _Float16* outw_h = (_Float16*)carve((size_t)CCH * DIN * 2);
  _Float16* f1w_h  = (_Float16*)carve((size_t)C2 * CCH * 2);
  _Float16* f3w_h  = (_Float16*)carve((size_t)CCH * C2 * 2);
  _Float16* xz_h0  = (_Float16*)carve((size_t)ROWS * NXZ * 2);
  _Float16* xz_h1  = (_Float16*)carve((size_t)ROWS * NXZ * 2);
  _Float16* xc_h0  = (_Float16*)carve((size_t)ROWS * DIN * 2);
  _Float16* xc_h1  = (_Float16*)carve((size_t)ROWS * DIN * 2);
  float*    dbl0   = (float*)   carve((size_t)ROWS * NDBL * 4);
  float*    dbl1   = (float*)   carve((size_t)ROWS * NDBL * 4);
  float*    yg0    = (float*)   carve((size_t)ROWS * DIN * 4);
  float*    yg1    = (float*)   carve((size_t)ROWS * DIN * 4);
  float*    deltab = (float*)   carve((size_t)ROWS * DIN * 4);      // shared by branches
  float*    Pc     = (float*)   carve((size_t)NCHUNK * BB * DIN * DSTATE * 4);
  float*    Sc     = (float*)   carve((size_t)NCHUNK * BB * DIN * DSTATE * 4);
  _Float16* ysum_h = (_Float16*)carve((size_t)ROWS * DIN * 2);
  float*    spat   = (float*)   carve((size_t)ROWS * CCH * 4);
  _Float16* ln_h   = (_Float16*)carve((size_t)ROWS * CCH * 2);
  float*    h1     = (float*)   carve((size_t)ROWS * C2 * 4);
  _Float16* h2_h   = (_Float16*)carve((size_t)ROWS * C2 * 2);

  float* out = (float*)d_out;                       // [h (B,C,64,64) | res (B,L,C)]

  // 1) residual add + layernorms (+ flip of e)
  k_add_ln<<<ROWS, CCH, 0, stream>>>(rgb, residual, depth, sn1w, sn1b, sn2w, sn2b,
                                     res, u_h, e_h);

  // shared weight conversions
  k_tohalf<<<(CCH*DIN + 255)/256, 256, 0, stream>>>(out_w, outw_h, CCH*DIN);
  k_tohalf<<<(C2*CCH + 255)/256, 256, 0, stream>>>(f1w, f1w_h, C2*CCH);
  k_tohalf<<<(CCH*C2 + 255)/256, 256, 0, stream>>>(f3w, f3w_h, CCH*C2);

  // 2) both mamba branches (sequential; scan scratch buffers are reused)
  for (int br = 0; br < 2; ++br) {
    int base = 17 + br * 8;
    const float* in_w   = (const float*)d_in[base + 0];
    const float* conv_w = (const float*)d_in[base + 1];
    const float* conv_b = (const float*)d_in[base + 2];
    const float* xproj  = (const float*)d_in[base + 3];
    const float* dt_w   = (const float*)d_in[base + 4];
    const float* dt_b   = (const float*)d_in[base + 5];
    const float* A_log  = (const float*)d_in[base + 6];
    const float* Dp     = (const float*)d_in[base + 7];
    _Float16* inw_h = br ? inw_h1 : inw_h0;
    _Float16* xz_h  = br ? xz_h1  : xz_h0;
    _Float16* xc_h  = br ? xc_h1  : xc_h0;
    float*    dbl   = br ? dbl1   : dbl0;
    float*    yg    = br ? yg1    : yg0;
    const _Float16* act = br ? e_h : u_h;

    k_tohalf<<<(NXZ*CCH + 255)/256, 256, 0, stream>>>(in_w, inw_h, NXZ*CCH);
    k_gemm_xz<<<(ROWS/128)*(NXZ/16), 256, 0, stream>>>(act, inw_h, xz_h);
    k_dwconv<<<(ROWS*DIN + 255)/256, 256, 0, stream>>>(xz_h, conv_w, conv_b, xc_h);
    k_dbl<<<(ROWS*NDBL + 255)/256, 256, 0, stream>>>(xc_h, xproj, dbl);
    k_delta<<<(ROWS*DIN + 255)/256, 256, 0, stream>>>(dbl, dt_w, dt_b, deltab);

    // chunk-parallel selective scan
    k_scan_p1<<<dim3(BB, NCHUNK), DIN, 0, stream>>>(dbl, deltab, xc_h, A_log, Pc, Sc);
    k_scan_p2<<<(BB*DIN*DSTATE + 255)/256, 256, 0, stream>>>(Pc, Sc);
    k_scan_p3<<<dim3(BB, NCHUNK), DIN, 0, stream>>>(dbl, deltab, xc_h, xz_h, A_log, Dp, Sc, yg);
  }

  // 3) combine branches, out-projection (+res, NCHW), LN2d
  k_ysum<<<(ROWS*DIN + 255)/256, 256, 0, stream>>>(yg0, yg1, ysum_h);
  k_gemm_gf<<<(ROWS/128)*(CCH/16), 256, 0, stream>>>(ysum_h, outw_h, res, spat);
  k_ln2d<<<ROWS, CCH, 0, stream>>>(spat, cnw, cnb, ln_h);

  // 4) conv stack: 1x1 (WMMA) -> grouped 3x3 -> 1x1 (WMMA, scatter to d_out)
  k_gemm_f1<<<(ROWS/128)*(C2/16), 256, 0, stream>>>(ln_h, f1w_h, f1b, h1);
  k_gconv<<<(ROWS*C2 + 255)/256, 256, 0, stream>>>(h1, f2w, f2b, h2_h);
  k_gemm_f3<<<(ROWS/128)*(CCH/16), 256, 0, stream>>>(h2_h, f3w_h, f3b, out);

  // 5) second output: res
  k_copy<<<(ROWS*CCH + 255)/256, 256, 0, stream>>>(res, out + (size_t)ROWS*CCH, ROWS*CCH);
}
